// WindowTransformer_38646115729660
// MI455X (gfx1250) — matrix operand order repaired and checked by reading, not compiled, not run on hardware
//
#include <hip/hip_runtime.h>
#include <hip/hip_bf16.h>
#include <math.h>

// ---------------------------------------------------------------------------
// MI455X (gfx1250) Swin window-attention block, f16 WMMA everywhere.
// v2: B-fragments (weights) hoisted across M-tiles (4x less L2 traffic, A
//     re-read from LDS instead), global_prefetch of next weight tile.
// ---------------------------------------------------------------------------

typedef __attribute__((ext_vector_type(16))) _Float16 v16h;
typedef __attribute__((ext_vector_type(8)))  _Float16 v8h;
typedef __attribute__((ext_vector_type(8)))  float    v8f;

#define DIM   256
#define HEADS 8
#define HD    32
#define NTOK  64
#define HID   1024
#define EPSLN 1e-5f
#define QSCALE 0.17677669529663687f   // 32^-0.5

// LDS strides (in halves); all row strides are multiples of 8 halves (16B).
#define XH_STRIDE 264   // 64 x 264  normalized input / attn-out tile
#define QK_STRIDE 40    // 64 x 40   q and k per head
#define VT_STRIDE 80    // 32 x 80   v transposed per head
#define PR_STRIDE 80    // 64 x 80   probs (overlays q+k region exactly)
#define HB_STRIDE 1040  // 64 x 1040 mlp hidden

__device__ __forceinline__ v8f wmma16(v16h a, v16h b, v8f c) {
    // v_wmma_f32_16x16x32_f16  (neg_a, A, neg_b, B, c_mod, C, reuse_a, reuse_b)
    return __builtin_amdgcn_wmma_f32_16x16x32_f16(false, a, false, b,
                                                  (short)0, c, false, false);
}

// A fragment, 16x32 f16. lane<16: M=lane, K={0..7,16..23}; lane>=16: K={8..15,24..31}
__device__ __forceinline__ v16h loadA(const _Float16* base, int stride,
                                      int m0, int k0, int lane) {
    int l = lane & 15, hi = lane >> 4;
    const _Float16* p = base + (size_t)(m0 + l) * stride + k0 + hi * 8;
    v8h lo = *(const v8h*)p;
    v8h hv = *(const v8h*)(p + 16);
    v16h r;
#pragma unroll
    for (int j = 0; j < 8; ++j) { r[j] = lo[j]; r[8 + j] = hv[j]; }
    return r;
}

// B fragment, 32x16 f16 from an [N][K]-major array (weights stored [out][in],
// K stored [token][ch], V stored [ch][token]). lane%16 selects column N,
// lane half selects K {0..7,16..23} / {8..15,24..31}.
__device__ __forceinline__ v16h loadB(const _Float16* base, int ldk,
                                      int n0, int k0, int lane) {
    int l = lane & 15, hi = lane >> 4;
    const _Float16* p = base + (size_t)(n0 + l) * ldk + k0 + hi * 8;
    v8h lo = *(const v8h*)p;
    v8h hv = *(const v8h*)(p + 16);
    v16h r;
#pragma unroll
    for (int j = 0; j < 8; ++j) { r[j] = lo[j]; r[8 + j] = hv[j]; }
    return r;
}

// ---------------------------------------------------------------------------
// Kernel 0: convert the four weight matrices f32 -> f16 into workspace.
// ---------------------------------------------------------------------------
__global__ __launch_bounds__(256)
void k_prep(const float* __restrict__ wqkv, const float* __restrict__ wproj,
            const float* __restrict__ wfc1, const float* __restrict__ wfc2,
            _Float16* __restrict__ oqkv, _Float16* __restrict__ oproj,
            _Float16* __restrict__ ofc1, _Float16* __restrict__ ofc2) {
    int i = blockIdx.x * 256 + threadIdx.x;
    if (i < 196608)       oqkv[i]          = (_Float16)wqkv[i];
    else if (i < 262144)  oproj[i-196608]  = (_Float16)wproj[i-196608];
    else if (i < 524288)  ofc1[i-262144]   = (_Float16)wfc1[i-262144];
    else if (i < 786432)  ofc2[i-524288]   = (_Float16)wfc2[i-524288];
}

// ---------------------------------------------------------------------------
// Kernel 1: LN1 + window gather + QKV + attention + proj + residual -> x1.
// One block per window; wave h owns head h end-to-end.
// ---------------------------------------------------------------------------
__global__ __launch_bounds__(256)
void k_attn(const float* __restrict__ x,
            const float* __restrict__ g1, const float* __restrict__ b1,
            const _Float16* __restrict__ wqkv, const float* __restrict__ bqkv,
            const float* __restrict__ btab,
            const _Float16* __restrict__ wproj, const float* __restrict__ bproj,
            float* __restrict__ x1) {
    extern __shared__ char smem[];
    _Float16* xh    = (_Float16*)smem;                       // 64 x 264
    _Float16* heads = xh + NTOK * XH_STRIDE;                 // 8 x (2*64*40 + 32*80)

    const int tid  = threadIdx.x;
    const int lane = tid & 31;
    const int wave = tid >> 5;            // == head index
    const int win  = blockIdx.x;
    const int b    = win >> 6;
    const int wy   = (win >> 3) & 7;
    const int wx   = win & 7;

    // ---- LayerNorm1 + window gather (4 lanes per token, 2-pass) ----
    {
        const int tok = tid >> 2, part = tid & 3;
        const int ty = tok >> 3, tx = tok & 7;
        const size_t grow = (size_t)b * 4096 + (size_t)(wy * 8 + ty) * 64 + (wx * 8 + tx);
        const float* xp = x + grow * DIM + part * 64;
        float s = 0.f, sq = 0.f;
#pragma unroll
        for (int j = 0; j < 64; j += 4) {
            float4 f = *(const float4*)(xp + j);
            s  += f.x + f.y + f.z + f.w;
            sq += f.x*f.x + f.y*f.y + f.z*f.z + f.w*f.w;
        }
        s  += __shfl_xor(s, 1, 32);  s  += __shfl_xor(s, 2, 32);
        sq += __shfl_xor(sq, 1, 32); sq += __shfl_xor(sq, 2, 32);
        const float mean = s * (1.f / 256.f);
        const float var  = sq * (1.f / 256.f) - mean * mean;
        const float rstd = rsqrtf(var + EPSLN);
        _Float16* dst = xh + (size_t)tok * XH_STRIDE + part * 64;
#pragma unroll
        for (int j = 0; j < 64; j += 4) {
            float4 f = *(const float4*)(xp + j);
            const int c = part * 64 + j;
            dst[j+0] = (_Float16)((f.x - mean) * rstd * g1[c+0] + b1[c+0]);
            dst[j+1] = (_Float16)((f.y - mean) * rstd * g1[c+1] + b1[c+1]);
            dst[j+2] = (_Float16)((f.z - mean) * rstd * g1[c+2] + b1[c+2]);
            dst[j+3] = (_Float16)((f.w - mean) * rstd * g1[c+3] + b1[c+3]);
        }
    }
    __syncthreads();

    const int h = wave;
    _Float16* qh = heads + (size_t)h * (2 * NTOK * QK_STRIDE + HD * VT_STRIDE);
    _Float16* kh = qh + NTOK * QK_STRIDE;
    _Float16* vt = kh + NTOK * QK_STRIDE;
    const int l  = lane & 15;
    const int hi = lane >> 4;

    // ---- QKV GEMM: wave h computes head h's q,k,v (64 x 32 each).
    //      B (weights) hoisted across all 4 M-tiles; A re-read from LDS. ----
#pragma unroll 1
    for (int pn = 0; pn < 6; ++pn) {            // piece*2 + nt
        const int piece = pn >> 1, nt = pn & 1;
        const int ncol  = piece * DIM + h * HD + nt * 16;
        // prefetch next output-tile's weight rows (per-lane: one row each)
        if (pn < 5) {
            const int pnn = pn + 1;
            const int ncn = (pnn >> 1) * DIM + h * HD + (pnn & 1) * 16;
            __builtin_prefetch(wqkv + (size_t)(ncn + l) * DIM, 0, 1);
        }
        v16h B[8];
#pragma unroll
        for (int kt = 0; kt < 8; ++kt)
            B[kt] = loadB(wqkv, DIM, ncol, kt * 32, lane);
        const float bia = bqkv[ncol + l];
#pragma unroll 1
        for (int mt = 0; mt < 4; ++mt) {
            v8f d = {};
#pragma unroll
            for (int kt = 0; kt < 8; ++kt)
                d = wmma16(loadA(xh, XH_STRIDE, mt * 16, kt * 32, lane), B[kt], d);
            if (piece == 0) {
#pragma unroll
                for (int v = 0; v < 8; ++v)
                    qh[(size_t)(mt*16 + v + 8*hi) * QK_STRIDE + nt*16 + l] =
                        (_Float16)((d[v] + bia) * QSCALE);
            } else if (piece == 1) {
#pragma unroll
                for (int v = 0; v < 8; ++v)
                    kh[(size_t)(mt*16 + v + 8*hi) * QK_STRIDE + nt*16 + l] =
                        (_Float16)(d[v] + bia);
            } else {  // v stored transposed: vt[ch][tok]
#pragma unroll
                for (int v = 0; v < 8; ++v)
                    vt[(size_t)(nt*16 + l) * VT_STRIDE + mt*16 + v + 8*hi] =
                        (_Float16)(d[v] + bia);
            }
        }
    }

    // ---- Attention (d=32 -> QK^T is ONE wmma per 16x16 tile) ----
    v16h QA[4], KB[4], VB[2][2];
#pragma unroll
    for (int mt = 0; mt < 4; ++mt) QA[mt] = loadA(qh, QK_STRIDE, mt * 16, 0, lane);
#pragma unroll
    for (int nt = 0; nt < 4; ++nt) KB[nt] = loadB(kh, QK_STRIDE, nt * 16, 0, lane);
#pragma unroll
    for (int kt = 0; kt < 2; ++kt)
#pragma unroll
        for (int nt = 0; nt < 2; ++nt)
            VB[kt][nt] = loadB(vt, VT_STRIDE, nt * 16, kt * 32, lane);

    _Float16* pr = qh;   // probs overlay q+k region: 64 x 80 halves exactly

#pragma unroll 1
    for (int mt = 0; mt < 4; ++mt) {
        v8f d[4];
#pragma unroll
        for (int nt = 0; nt < 4; ++nt) { v8f z = {}; d[nt] = wmma16(QA[mt], KB[nt], z); }
        // relative position bias
#pragma unroll
        for (int nt = 0; nt < 4; ++nt)
#pragma unroll
            for (int v = 0; v < 8; ++v) {
                const int m = mt * 16 + v + 8 * hi;
                const int n = nt * 16 + l;
                const int idx = ((m >> 3) - (n >> 3) + 7) * 15 + ((m & 7) - (n & 7) + 7);
                d[nt][v] += btab[idx * HEADS + h];
            }
        // row softmax (rows live across 16-lane halves)
#pragma unroll
        for (int v = 0; v < 8; ++v) {
            float mx = fmaxf(fmaxf(d[0][v], d[1][v]), fmaxf(d[2][v], d[3][v]));
            mx = fmaxf(mx, __shfl_xor(mx, 1, 32));
            mx = fmaxf(mx, __shfl_xor(mx, 2, 32));
            mx = fmaxf(mx, __shfl_xor(mx, 4, 32));
            mx = fmaxf(mx, __shfl_xor(mx, 8, 32));
            float s = 0.f;
#pragma unroll
            for (int nt = 0; nt < 4; ++nt) { d[nt][v] = __expf(d[nt][v] - mx); s += d[nt][v]; }
            s += __shfl_xor(s, 1, 32); s += __shfl_xor(s, 2, 32);
            s += __shfl_xor(s, 4, 32); s += __shfl_xor(s, 8, 32);
            const float inv = 1.f / s;
#pragma unroll
            for (int nt = 0; nt < 4; ++nt) d[nt][v] *= inv;
        }
#pragma unroll
        for (int nt = 0; nt < 4; ++nt)
#pragma unroll
            for (int v = 0; v < 8; ++v)
                pr[(size_t)(mt*16 + v + 8*hi) * PR_STRIDE + nt*16 + l] = (_Float16)d[nt][v];
    }

    // ---- attn @ V -> per-head output (64 x 32) ----
    v8f o[4][2];
#pragma unroll 1
    for (int mt = 0; mt < 4; ++mt)
#pragma unroll
        for (int nt = 0; nt < 2; ++nt) {
            v8f acc = {};
#pragma unroll
            for (int kt = 0; kt < 2; ++kt)
                acc = wmma16(loadA(pr, PR_STRIDE, mt * 16, kt * 32, lane), VB[kt][nt], acc);
            o[mt][nt] = acc;
        }

    __syncthreads();   // all waves done reading xh -> safe to overwrite
#pragma unroll
    for (int mt = 0; mt < 4; ++mt)
#pragma unroll
        for (int nt = 0; nt < 2; ++nt)
#pragma unroll
            for (int v = 0; v < 8; ++v)
                xh[(size_t)(mt*16 + v + 8*hi) * XH_STRIDE + h*32 + nt*16 + l] =
                    (_Float16)o[mt][nt][v];
    __syncthreads();

    // ---- proj GEMM + shortcut residual -> x1 (B hoisted across M-tiles) ----
#pragma unroll 1
    for (int i = 0; i < 2; ++i) {
        const int nt = wave * 2 + i;
        if (i == 0)
            __builtin_prefetch(wproj + (size_t)((nt + 1) * 16 + l) * DIM, 0, 1);
        v16h B[8];
#pragma unroll
        for (int kt = 0; kt < 8; ++kt)
            B[kt] = loadB(wproj, DIM, nt * 16, kt * 32, lane);
        const float bp = bproj[nt * 16 + l];
#pragma unroll 1
        for (int mt = 0; mt < 4; ++mt) {
            v8f d = {};
#pragma unroll
            for (int kt = 0; kt < 8; ++kt)
                d = wmma16(loadA(xh, XH_STRIDE, mt * 16, kt * 32, lane), B[kt], d);
#pragma unroll
            for (int v = 0; v < 8; ++v) {
                const int m = mt * 16 + v + 8 * hi;
                const int ty = m >> 3, tx = m & 7;
                const size_t grow = (size_t)b * 4096 + (size_t)(wy * 8 + ty) * 64 + (wx * 8 + tx);
                const size_t off = grow * DIM + nt * 16 + l;
                x1[off] = x[off] + d[v] + bp;
            }
        }
    }
}

// ---------------------------------------------------------------------------
// Kernel 2: LN2 + FC1 + GELU + FC2 + residual -> out. One block per 64 tokens.
// ---------------------------------------------------------------------------
__global__ __launch_bounds__(256)
void k_mlp(const float* __restrict__ x1,
           const float* __restrict__ g2, const float* __restrict__ b2,
           const _Float16* __restrict__ wfc1, const float* __restrict__ bfc1,
           const _Float16* __restrict__ wfc2, const float* __restrict__ bfc2,
           float* __restrict__ out) {
    extern __shared__ char smem[];
    _Float16* xh = (_Float16*)smem;            // 64 x 264
    _Float16* hb = xh + NTOK * XH_STRIDE;      // 64 x 1040

    const int tid  = threadIdx.x;
    const int lane = tid & 31;
    const int wave = tid >> 5;
    const int l    = lane & 15;
    const int hi   = lane >> 4;
    const size_t row0 = (size_t)blockIdx.x * 64;

    // ---- LayerNorm2 ----
    {
        const int tok = tid >> 2, part = tid & 3;
        const float* xp = x1 + (row0 + tok) * DIM + part * 64;
        float s = 0.f, sq = 0.f;
#pragma unroll
        for (int j = 0; j < 64; j += 4) {
            float4 f = *(const float4*)(xp + j);
            s  += f.x + f.y + f.z + f.w;
            sq += f.x*f.x + f.y*f.y + f.z*f.z + f.w*f.w;
        }
        s  += __shfl_xor(s, 1, 32);  s  += __shfl_xor(s, 2, 32);
        sq += __shfl_xor(sq, 1, 32); sq += __shfl_xor(sq, 2, 32);
        const float mean = s * (1.f / 256.f);
        const float rstd = rsqrtf(sq * (1.f / 256.f) - mean * mean + EPSLN);
        _Float16* dst = xh + (size_t)tok * XH_STRIDE + part * 64;
#pragma unroll
        for (int j = 0; j < 64; j += 4) {
            float4 f = *(const float4*)(xp + j);
            const int c = part * 64 + j;
            dst[j+0] = (_Float16)((f.x - mean) * rstd * g2[c+0] + b2[c+0]);
            dst[j+1] = (_Float16)((f.y - mean) * rstd * g2[c+1] + b2[c+1]);
            dst[j+2] = (_Float16)((f.z - mean) * rstd * g2[c+2] + b2[c+2]);
            dst[j+3] = (_Float16)((f.w - mean) * rstd * g2[c+3] + b2[c+3]);
        }
    }
    __syncthreads();

    // ---- FC1 + exact GELU -> hidden (f16 in LDS).
    //      B hoisted across M-tiles; A re-read from LDS; prefetch next tile. ----
#pragma unroll 1
    for (int i = 0; i < 8; ++i) {
        const int nt = wave * 8 + i;
        if (i < 7)
            __builtin_prefetch(wfc1 + (size_t)((nt + 1) * 16 + l) * DIM, 0, 1);
        v16h B[8];
#pragma unroll
        for (int kt = 0; kt < 8; ++kt)
            B[kt] = loadB(wfc1, DIM, nt * 16, kt * 32, lane);
        const float bb = bfc1[nt * 16 + l];
#pragma unroll 1
        for (int mt = 0; mt < 4; ++mt) {
            v8f d = {};
#pragma unroll
            for (int kt = 0; kt < 8; ++kt)
                d = wmma16(loadA(xh, XH_STRIDE, mt * 16, kt * 32, lane), B[kt], d);
#pragma unroll
            for (int v = 0; v < 8; ++v) {
                float t = d[v] + bb;
                t = 0.5f * t * (1.f + erff(t * 0.70710678118654752f));
                hb[(size_t)(mt*16 + v + 8*hi) * HB_STRIDE + nt*16 + l] = (_Float16)t;
            }
        }
    }
    __syncthreads();

    // ---- FC2 + residual. K=1024: tile kt in blocks of 8 with 4 live
    //      accumulators so B is still hoisted across all M-tiles. ----
#pragma unroll 1
    for (int i = 0; i < 2; ++i) {
        const int nt = wave * 2 + i;
        v8f acc[4] = {{}, {}, {}, {}};
#pragma unroll 1
        for (int ktb = 0; ktb < 4; ++ktb) {
            if (ktb < 3)
                __builtin_prefetch(wfc2 + (size_t)(nt * 16 + l) * HID + (ktb + 1) * 256, 0, 1);
            v16h B[8];
#pragma unroll
            for (int kt = 0; kt < 8; ++kt)
                B[kt] = loadB(wfc2, HID, nt * 16, ktb * 256 + kt * 32, lane);
#pragma unroll 1
            for (int mt = 0; mt < 4; ++mt) {
#pragma unroll
                for (int kt = 0; kt < 8; ++kt)
                    acc[mt] = wmma16(loadA(hb, HB_STRIDE, mt * 16, ktb * 256 + kt * 32, lane),
                                     B[kt], acc[mt]);
            }
        }
        const float bb = bfc2[nt * 16 + l];
#pragma unroll
        for (int mt = 0; mt < 4; ++mt)
#pragma unroll
            for (int v = 0; v < 8; ++v) {
                const size_t off = (row0 + mt*16 + v + 8*hi) * DIM + nt*16 + l;
                out[off] = x1[off] + acc[mt][v] + bb;
            }
    }
}

// ---------------------------------------------------------------------------
extern "C" void kernel_launch(void* const* d_in, const int* in_sizes, int n_in,
                              void* d_out, int out_size, void* d_ws, size_t ws_size,
                              hipStream_t stream) {
    const float* x     = (const float*)d_in[0];
    // d_in[1] attn_mask: unused (reference passes mask=None)
    const float* g1    = (const float*)d_in[2];
    const float* b1    = (const float*)d_in[3];
    const float* wqkv  = (const float*)d_in[4];
    const float* bqkv  = (const float*)d_in[5];
    const float* btab  = (const float*)d_in[6];
    const float* wproj = (const float*)d_in[7];
    const float* bproj = (const float*)d_in[8];
    const float* g2    = (const float*)d_in[9];
    const float* b2    = (const float*)d_in[10];
    const float* wfc1  = (const float*)d_in[11];
    const float* bfc1  = (const float*)d_in[12];
    const float* wfc2  = (const float*)d_in[13];
    const float* bfc2  = (const float*)d_in[14];
    float* out = (float*)d_out;

    char* ws = (char*)d_ws;
    float* x1 = (float*)ws;                      // 32*4096*256 f32 = 128 MiB
    size_t off = (size_t)32 * 4096 * 256 * 4;
    _Float16* hqkv  = (_Float16*)(ws + off); off += (size_t)196608 * 2;
    _Float16* hproj = (_Float16*)(ws + off); off += (size_t)65536 * 2;
    _Float16* hfc1  = (_Float16*)(ws + off); off += (size_t)262144 * 2;
    _Float16* hfc2  = (_Float16*)(ws + off);

    const size_t lds_attn = (size_t)(NTOK * XH_STRIDE +
                            HEADS * (2 * NTOK * QK_STRIDE + HD * VT_STRIDE)) * 2; // 156672 B
    const size_t lds_mlp  = (size_t)(NTOK * XH_STRIDE + NTOK * HB_STRIDE) * 2;    // 166912 B

    hipFuncSetAttribute((const void*)k_attn,
                        hipFuncAttributeMaxDynamicSharedMemorySize, (int)lds_attn);
    hipFuncSetAttribute((const void*)k_mlp,
                        hipFuncAttributeMaxDynamicSharedMemorySize, (int)lds_mlp);

    k_prep<<<3072, 256, 0, stream>>>(wqkv, wproj, wfc1, wfc2, hqkv, hproj, hfc1, hfc2);
    k_attn<<<2048, 256, lds_attn, stream>>>(x, g1, b1, hqkv, bqkv, btab, hproj, bproj, x1);
    k_mlp<<<2048, 256, lds_mlp, stream>>>(x1, g2, b2, hfc1, bfc1, hfc2, bfc2, out);
}
